// MultiHeadAttention_88656714925519
// MI455X (gfx1250) — matrix operand order repaired and checked by reading, not compiled, not run on hardware
//
#include <hip/hip_runtime.h>

// MI455X / gfx1250, wave32. bf16 WMMA path: v_wmma_f32_16x16x32_bf16.

typedef __attribute__((ext_vector_type(16))) __bf16 v16bf;
typedef __attribute__((ext_vector_type(8)))  __bf16 v8bf;
typedef __attribute__((ext_vector_type(8)))  float  v8f;
typedef __attribute__((ext_vector_type(2)))  __bf16 bf2;
typedef __attribute__((ext_vector_type(4)))  float  f4;

#define DMODEL 1024
#define NTOK   2048
#define NHEAD  16
#define DKH    64
#define NEGINF (-1.0e9f)

// A-matrix (16x32 bf16) per-lane K offset for VGPR pair v (contiguous order):
// VGPR v holds K = half*16 + v*2 (+1)
__device__ __forceinline__ int a_koff(int v, int half) {
  return (half << 4) + (v << 1);
}

__device__ __forceinline__ v8f wmma_bf16(v16bf a, v16bf b, v8f c) {
  return __builtin_amdgcn_wmma_f32_16x16x32_bf16(
      /*neg_a=*/false, a, /*neg_b=*/false, b,
      /*c_mod=*/(short)0, c, /*reuse_a=*/false, /*reuse_b=*/false);
}

// ---------------------------------------------------------------------------
// Elementwise f32 -> bf16 (one-time precision downconvert; 8 elts/thread)
// ---------------------------------------------------------------------------
__global__ __launch_bounds__(256, 1) void cvt_f32_to_bf16(
    const float* __restrict__ src, __bf16* __restrict__ dst, int n)
{
  const int i = (blockIdx.x * 256 + threadIdx.x) * 8;
  if (i + 8 <= n) {
    f4 p0 = *(const f4*)(src + i);
    f4 p1 = *(const f4*)(src + i + 4);
    v8bf o;
    o[0] = (__bf16)p0.x; o[1] = (__bf16)p0.y; o[2] = (__bf16)p0.z; o[3] = (__bf16)p0.w;
    o[4] = (__bf16)p1.x; o[5] = (__bf16)p1.y; o[6] = (__bf16)p1.z; o[7] = (__bf16)p1.w;
    *(v8bf*)(dst + i) = o;
  }
}

// ---------------------------------------------------------------------------
// Y[M=2048, N=1024] = (X[2048,1024] @ W^T + bias) * scale, bf16 in, 2x2
// register-blocked: each wave computes a 32x32 output tile (4 WMMAs/K-step
// sharing 2 A + 2 B fragments). Block = 8 waves (2 M x 4 N) = 64x128 tile.
// OUT_MODE: 0 = f32 row-major, 1 = bf16 row-major, 2 = bf16 transposed [N][NTOK]
// ---------------------------------------------------------------------------
template <int OUT_MODE>
__global__ __launch_bounds__(256, 1) void proj_gemm(
    const __bf16* __restrict__ X,
    const __bf16* __restrict__ W,
    const float*  __restrict__ bias,
    float scale,
    float*  __restrict__ Yf,
    __bf16* __restrict__ Yb)
{
  const int tid  = threadIdx.x;
  const int wave = tid >> 5;
  const int lane = tid & 31;
  const int half = lane >> 4;
  const int l16  = lane & 15;
  const int m0 = (blockIdx.y << 6) + ((wave >> 2) << 5);
  const int n0 = (blockIdx.x << 7) + ((wave & 3) << 5);

  const __bf16* xrow0 = X + (size_t)(m0 + l16) * DMODEL;
  const __bf16* xrow1 = X + (size_t)(m0 + 16 + l16) * DMODEL;
  const __bf16* wrow0 = W + (size_t)(n0 + l16) * DMODEL;
  const __bf16* wrow1 = W + (size_t)(n0 + 16 + l16) * DMODEL;

  v8f acc00 = {}, acc01 = {}, acc10 = {}, acc11 = {};

  #pragma unroll 2
  for (int kk = 0; kk < DMODEL; kk += 32) {
    __builtin_prefetch(wrow0 + kk + 256, 0, 0);   // global_prefetch_b8
    __builtin_prefetch(wrow1 + kk + 256, 0, 0);
    v16bf a0, a1, b0, b1;
    #pragma unroll
    for (int v = 0; v < 8; ++v) {
      const int ko = a_koff(v, half);
      bf2 pa0 = *(const bf2*)(xrow0 + kk + ko);
      bf2 pa1 = *(const bf2*)(xrow1 + kk + ko);
      a0[2 * v] = pa0.x; a0[2 * v + 1] = pa0.y;
      a1[2 * v] = pa1.x; a1[2 * v + 1] = pa1.y;
      // B 32x16: lanes 0-15 K=kk+0..15, lanes 16-31 K=kk+16..31
      const int kp = kk + (half << 4) + (v << 1);
      bf2 pb0 = *(const bf2*)(wrow0 + kp);
      bf2 pb1 = *(const bf2*)(wrow1 + kp);
      b0[2 * v] = pb0.x; b0[2 * v + 1] = pb0.y;
      b1[2 * v] = pb1.x; b1[2 * v + 1] = pb1.y;
    }
    acc00 = wmma_bf16(a0, b0, acc00);
    acc01 = wmma_bf16(a0, b1, acc01);
    acc10 = wmma_bf16(a1, b0, acc10);
    acc11 = wmma_bf16(a1, b1, acc11);
  }

  const float bs0 = bias[n0 + l16];
  const float bs1 = bias[n0 + 16 + l16];
  const int col0 = n0 + l16;
  const int col1 = n0 + 16 + l16;
  #pragma unroll
  for (int mi = 0; mi < 2; ++mi) {
    const v8f c0 = mi ? acc10 : acc00;
    const v8f c1 = mi ? acc11 : acc01;
    #pragma unroll
    for (int r = 0; r < 8; ++r) {
      const int row = m0 + (mi << 4) + (half << 3) + r;   // C layout row
      const float v0 = (c0[r] + bs0) * scale;
      const float v1 = (c1[r] + bs1) * scale;
      if (OUT_MODE == 0) {
        Yf[(size_t)row * DMODEL + col0] = v0;
        Yf[(size_t)row * DMODEL + col1] = v1;
      } else if (OUT_MODE == 1) {
        Yb[(size_t)row * DMODEL + col0] = (__bf16)v0;
        Yb[(size_t)row * DMODEL + col1] = (__bf16)v1;
      } else {  // transposed: vT[n][token] -> contiguous K-pairs for PV WMMA
        Yb[(size_t)col0 * NTOK + row] = (__bf16)v0;
        Yb[(size_t)col1 * NTOK + row] = (__bf16)v1;
      }
    }
  }
}

// ---------------------------------------------------------------------------
// Fused attention: block = (head h, 16-query-row tile m0). 8 waves, each wave
// owns a 256-wide key strip. scores -> mask -> softmax -> p_attn (f32 global)
// + P (bf16, LDS transpose stage) -> P@V (vT layout) -> cross-wave reduce -> xb.
// ---------------------------------------------------------------------------
__global__ __launch_bounds__(256, 1) void attn_fused(
    const __bf16* __restrict__ qb,
    const __bf16* __restrict__ kb,
    const __bf16* __restrict__ vT,  // [1024][2048]: vT[h*64+d][token]
    const int*    __restrict__ mask,
    float*  __restrict__ p_attn,    // [16, 2048, 2048]
    __bf16* __restrict__ xb)        // [2048, 1024]
{
  __shared__ __align__(16) __bf16 pstage[8][16 * 32]; // per-wave P transpose stage
  __shared__ float  xpart[8][16 * 64];                 // per-wave PV partials
  __shared__ float  redbuf[8][16];                     // per-wave row stats
  __shared__ float  rowstat[16];                       // block row stats

  const int tid   = threadIdx.x;
  const int wave  = tid >> 5;
  const int lane  = tid & 31;
  const int half  = lane >> 4;
  const int l16   = lane & 15;
  const int m0    = blockIdx.x << 4;
  const int h     = blockIdx.y;
  const int hq    = h * DKH;
  const int jbase = wave << 8;      // this wave's 256 key columns

  // ---- q A-fragments: 16x64 = two 16x32 (kept in registers) ----
  v16bf a0, a1;
  {
    const __bf16* qrow = qb + (size_t)(m0 + l16) * DMODEL + hq;
    #pragma unroll
    for (int v = 0; v < 8; ++v) {
      const int ko = a_koff(v, half);
      bf2 p0 = *(const bf2*)(qrow + ko);
      bf2 p1 = *(const bf2*)(qrow + 32 + ko);
      a0[2 * v] = p0.x; a0[2 * v + 1] = p0.y;
      a1[2 * v] = p1.x; a1[2 * v + 1] = p1.y;
    }
  }

  // ---- scores: 16 tiles of 16 keys, 2 WMMAs each (K=64) ----
  v8f s[16];
  #pragma unroll
  for (int t = 0; t < 16; ++t) {
    const int j = jbase + (t << 4) + l16;        // key index for this lane
    const __bf16* krow = kb + (size_t)j * DMODEL + hq;
    v16bf b0, b1;
    #pragma unroll
    for (int v = 0; v < 8; ++v) {
      const int d = (half << 4) + (v << 1);
      bf2 p0 = *(const bf2*)(krow + d);
      bf2 p1 = *(const bf2*)(krow + 32 + d);
      b0[2 * v] = p0.x; b0[2 * v + 1] = p0.y;
      b1[2 * v] = p1.x; b1[2 * v + 1] = p1.y;
    }
    v8f c = {};
    c = wmma_bf16(a0, b0, c);
    c = wmma_bf16(a1, b1, c);
    if (mask[j] == 0) {
      #pragma unroll
      for (int r = 0; r < 8; ++r) c[r] = NEGINF;
    }
    s[t] = c;
  }

  // ---- row max: lanes (16-group shfl) -> waves (LDS) ----
  float pmax[8];
  #pragma unroll
  for (int r = 0; r < 8; ++r) pmax[r] = NEGINF;
  #pragma unroll
  for (int t = 0; t < 16; ++t)
    #pragma unroll
    for (int r = 0; r < 8; ++r) pmax[r] = fmaxf(pmax[r], s[t][r]);
  #pragma unroll
  for (int off = 1; off < 16; off <<= 1)
    #pragma unroll
    for (int r = 0; r < 8; ++r)
      pmax[r] = fmaxf(pmax[r], __shfl_xor(pmax[r], off, 32));
  if (l16 == 0) {
    #pragma unroll
    for (int r = 0; r < 8; ++r) redbuf[wave][(half << 3) + r] = pmax[r];
  }
  __syncthreads();
  if (tid < 16) {
    float m = redbuf[0][tid];
    #pragma unroll
    for (int w = 1; w < 8; ++w) m = fmaxf(m, redbuf[w][tid]);
    rowstat[tid] = m;
  }
  __syncthreads();
  float rmax[8];
  #pragma unroll
  for (int r = 0; r < 8; ++r) rmax[r] = rowstat[(half << 3) + r];

  // ---- exp + row sum ----
  float psum[8] = {0.f, 0.f, 0.f, 0.f, 0.f, 0.f, 0.f, 0.f};
  #pragma unroll
  for (int t = 0; t < 16; ++t)
    #pragma unroll
    for (int r = 0; r < 8; ++r) {
      const float e = __expf(s[t][r] - rmax[r]);
      s[t][r] = e;
      psum[r] += e;
    }
  #pragma unroll
  for (int off = 1; off < 16; off <<= 1)
    #pragma unroll
    for (int r = 0; r < 8; ++r) psum[r] += __shfl_xor(psum[r], off, 32);
  if (l16 == 0) {
    #pragma unroll
    for (int r = 0; r < 8; ++r) redbuf[wave][(half << 3) + r] = psum[r];
  }
  __syncthreads();
  if (tid < 16) {
    float sm = redbuf[0][tid];
    #pragma unroll
    for (int w = 1; w < 8; ++w) sm += redbuf[w][tid];
    rowstat[tid] = sm;
  }
  __syncthreads();
  float rinv[8];
  #pragma unroll
  for (int r = 0; r < 8; ++r) rinv[r] = 1.0f / rowstat[(half << 3) + r];

  // ---- normalize, emit p_attn, stage P (bf16) in LDS, P@V WMMAs ----
  v8f acc[4] = {{}, {}, {}, {}};                 // 16 rows x 64 d partial
  float* prow = p_attn + ((size_t)h * NTOK + m0) * NTOK;

  #pragma unroll
  for (int u = 0; u < 8; ++u) {                  // two 16-col tiles per step
    asm volatile("s_wait_dscnt 0x0" ::: "memory");   // prior A-frag reads done
    #pragma unroll
    for (int sub = 0; sub < 2; ++sub) {
      const int t = 2 * u + sub;
      const int j = jbase + (t << 4) + l16;
      #pragma unroll
      for (int r = 0; r < 8; ++r) {
        const int row = (half << 3) + r;         // C layout row
        const float p = s[t][r] * rinv[r];
        prow[(size_t)row * NTOK + j] = p;
        pstage[wave][(row << 5) + (sub << 4) + l16] = (__bf16)p;
      }
    }
    asm volatile("s_wait_dscnt 0x0" ::: "memory");   // stage visible to wave
    // A fragment (16x32) of P from LDS, now in A layout
    v16bf ap;
    #pragma unroll
    for (int v = 0; v < 8; ++v) {
      const int ko = a_koff(v, half);
      bf2 p = *(const bf2*)(&pstage[wave][(l16 << 5) + ko]);
      ap[2 * v] = p.x; ap[2 * v + 1] = p.y;
    }
    // B fragments from vT: B(k=j, n=d) = vT[hq+d][j]; K-pairs contiguous
    const int kglob = jbase + (u << 5);
    #pragma unroll
    for (int nt = 0; nt < 4; ++nt) {
      const int d = (nt << 4) + l16;
      const __bf16* vrow = vT + (size_t)(hq + d) * NTOK + kglob + (half << 4);
      v16bf bfrag;
      #pragma unroll
      for (int v = 0; v < 8; ++v) {
        bf2 p = *(const bf2*)(vrow + (v << 1));
        bfrag[2 * v] = p.x; bfrag[2 * v + 1] = p.y;
      }
      acc[nt] = wmma_bf16(ap, bfrag, acc[nt]);
    }
  }

  // ---- cross-wave reduction of PV partials -> xb (bf16) ----
  #pragma unroll
  for (int nt = 0; nt < 4; ++nt)
    #pragma unroll
    for (int r = 0; r < 8; ++r)
      xpart[wave][((half << 3) + r) * 64 + (nt << 4) + l16] = acc[nt][r];
  __syncthreads();
  for (int idx = tid; idx < 16 * 64; idx += 256) {
    float sum = 0.f;
    #pragma unroll
    for (int w = 0; w < 8; ++w) sum += xpart[w][idx];
    const int row = idx >> 6;
    const int col = idx & 63;
    xb[(size_t)(m0 + row) * DMODEL + hq + col] = (__bf16)sum;
  }
}

// ---------------------------------------------------------------------------
extern "C" void kernel_launch(void* const* d_in, const int* in_sizes, int n_in,
                              void* d_out, int out_size, void* d_ws, size_t ws_size,
                              hipStream_t stream) {
  (void)in_sizes; (void)n_in; (void)out_size; (void)ws_size;

  const float* query = (const float*)d_in[0];
  const float* key_  = (const float*)d_in[1];
  const float* value = (const float*)d_in[2];
  const int*   mask  = (const int*)  d_in[3];
  const float* Wq = (const float*)d_in[4];
  const float* bq = (const float*)d_in[5];
  const float* Wk = (const float*)d_in[6];
  const float* bk = (const float*)d_in[7];
  const float* Wv = (const float*)d_in[8];
  const float* bv = (const float*)d_in[9];
  const float* Wo = (const float*)d_in[10];
  const float* bo = (const float*)d_in[11];

  float* out    = (float*)d_out;                 // [2048*1024]
  float* p_attn = out + (size_t)NTOK * DMODEL;   // [16*2048*2048]

  // workspace layout (36 MB total)
  const size_t ACT = (size_t)NTOK * DMODEL * sizeof(__bf16);   // 4 MB
  const size_t WSZ = (size_t)DMODEL * DMODEL * sizeof(__bf16); // 2 MB
  char* ws = (char*)d_ws;
  __bf16* qbuf = (__bf16*)(ws + 0 * ACT);
  __bf16* kbuf = (__bf16*)(ws + 1 * ACT);
  __bf16* vT   = (__bf16*)(ws + 2 * ACT);   // [1024][2048]
  __bf16* xbuf = (__bf16*)(ws + 3 * ACT);
  __bf16* Xq   = (__bf16*)(ws + 4 * ACT);
  __bf16* Xk   = (__bf16*)(ws + 5 * ACT);
  __bf16* Xv   = (__bf16*)(ws + 6 * ACT);
  __bf16* Wqb  = (__bf16*)(ws + 7 * ACT + 0 * WSZ);
  __bf16* Wkb  = (__bf16*)(ws + 7 * ACT + 1 * WSZ);
  __bf16* Wvb  = (__bf16*)(ws + 7 * ACT + 2 * WSZ);
  __bf16* Wob  = (__bf16*)(ws + 7 * ACT + 3 * WSZ);

  const int nAct = NTOK * DMODEL;      // 2M elts
  const int nW   = DMODEL * DMODEL;    // 1M elts
  cvt_f32_to_bf16<<<nAct / 2048, 256, 0, stream>>>(query, Xq, nAct);
  cvt_f32_to_bf16<<<nAct / 2048, 256, 0, stream>>>(key_,  Xk, nAct);
  cvt_f32_to_bf16<<<nAct / 2048, 256, 0, stream>>>(value, Xv, nAct);
  cvt_f32_to_bf16<<<nW / 2048, 256, 0, stream>>>(Wq, Wqb, nW);
  cvt_f32_to_bf16<<<nW / 2048, 256, 0, stream>>>(Wk, Wkb, nW);
  cvt_f32_to_bf16<<<nW / 2048, 256, 0, stream>>>(Wv, Wvb, nW);
  cvt_f32_to_bf16<<<nW / 2048, 256, 0, stream>>>(Wo, Wob, nW);

  dim3 gproj(DMODEL / 128, NTOK / 64);   // (8, 32), 256 threads
  // q scaled by 1/sqrt(d_k) = 1/8, folded into bf16 Q
  proj_gemm<1><<<gproj, 256, 0, stream>>>(Xq, Wqb, bq, 0.125f, nullptr, qbuf);
  proj_gemm<1><<<gproj, 256, 0, stream>>>(Xk, Wkb, bk, 1.0f,   nullptr, kbuf);
  proj_gemm<2><<<gproj, 256, 0, stream>>>(Xv, Wvb, bv, 1.0f,   nullptr, vT);

  dim3 gattn(NTOK / 16, NHEAD);          // (128, 16), 256 threads
  attn_fused<<<gattn, 256, 0, stream>>>(qbuf, kbuf, vT, mask, p_attn, xbuf);

  proj_gemm<0><<<gproj, 256, 0, stream>>>(xbuf, Wob, bo, 1.0f, out, nullptr);
}
